// UnifiedEquivariantHourglass_43224550867014
// MI455X (gfx1250) — compile-verified, matrix operand order checked by reading, not run on hardware
//
#include <hip/hip_runtime.h>
#include <hip/hip_bf16.h>

// ---------------------------------------------------------------------------
// CDNA5 (gfx1250): wave32 + v_wmma_f32_16x16x32_bf16 GEMMs.
// - Weights are pre-tiled once per call into bf16 *fragment order* tiles
//   (512 ushorts / 16x16x32 tile) in workspace: GEMM B-staging is a pure
//   16B-chunk copy -> GLOBAL_LOAD_ASYNC_TO_LDS_B128 (ASYNCcnt) when the
//   toolchain exposes the builtin, uint4 copy fallback otherwise.
// - Each WMMA operand is one 32B LDS read (2x ds_load_b128).
// - fp32->bf16 activation staging uses RNE + v_perm_b32 packing.
// ---------------------------------------------------------------------------

typedef __attribute__((ext_vector_type(16))) __bf16 v16bf;
typedef __attribute__((ext_vector_type(8)))  float  v8f;
typedef __attribute__((ext_vector_type(4)))  int    v4i;

__constant__ int c_fano[7][3] = {{0,1,2},{0,3,4},{0,5,6},{1,3,5},{1,4,6},{2,3,6},{2,4,5}};

#if __has_builtin(__builtin_amdgcn_global_load_async_to_lds_b128)
#define HAVE_ASYNC 1
#else
#define HAVE_ASYNC 0
#endif

// 16-byte global->LDS copy (async when available).
__device__ __forceinline__ void cp16(const unsigned short* g, unsigned short* l) {
#if HAVE_ASYNC
    __builtin_amdgcn_global_load_async_to_lds_b128((v4i*)g, (v4i*)l, 0, 0);
#else
    *(uint4*)l = *(const uint4*)g;
#endif
}
__device__ __forceinline__ void cp_wait() {
#if HAVE_ASYNC
#if __has_builtin(__builtin_amdgcn_s_wait_asynccnt)
    __builtin_amdgcn_s_wait_asynccnt(0);
#else
    asm volatile("s_wait_asynccnt 0x0" ::: "memory");
#endif
#endif
}

// Pack two fp32 into two RNE-rounded bf16 in one dword (v_perm_b32 pack).
__device__ __forceinline__ unsigned int pack2bf(float a, float b) {
    unsigned int ua = __float_as_uint(a);
    unsigned int ub = __float_as_uint(b);
    ua += 0x7FFFu + ((ua >> 16) & 1u);
    ub += 0x7FFFu + ((ub >> 16) & 1u);
    return __builtin_amdgcn_perm(ub, ua, 0x07060302u);
}

// Fragment-order tile: tile[lane*16 + e], 512 ushorts per tile.
//   lane = row(or col) + 16*((k>>3)&1),  e = (k&7) + 8*(k>>4)
__device__ __forceinline__ v16bf frag_ld(const unsigned short* tile) {
    return *(const v16bf*)(tile + (threadIdx.x & 31) * 16);
}

__device__ __forceinline__ void stage_a1x2(unsigned short* tile, int m, int k,
                                           float a, float b) {
    int lane = m + (((k >> 3) & 1) << 4);
    int e0 = (k & 7) + ((k >> 4) << 3);
    *(unsigned int*)(tile + lane * 16 + e0) = pack2bf(a, b);
}

__device__ __forceinline__ void stage_a1x4(unsigned short* tile, int m, int k,
                                           const float4& v) {
    int lane = m + (((k >> 3) & 1) << 4);
    int e0 = (k & 7) + ((k >> 4) << 3);
    *(uint2*)(tile + lane * 16 + e0) = make_uint2(pack2bf(v.x, v.y), pack2bf(v.z, v.w));
}

__device__ __forceinline__ v8f wmma_bf16(v16bf a, v16bf b, v8f c) {
    return __builtin_amdgcn_wmma_f32_16x16x32_bf16(
        false, a, false, b, (short)0, c, false, false);
}

// ---------------------------------------------------------------------------
// Kernel 0: pre-tile an fp32 [K][N] weight into bf16 fragment-order tiles.
// dst tile (kt,nt) occupies 512 ushorts at (kt*(N/16)+nt)*512.
// One output uint (two consecutive k for fixed n) per thread.
// ---------------------------------------------------------------------------
__global__ __launch_bounds__(256) void k_pretile(
    const float* __restrict__ W, unsigned short* __restrict__ dst,
    int N, int totalU)
{
    int u = blockIdx.x * 256 + threadIdx.x;
    if (u >= totalU) return;
    int tile = u >> 8, w = u & 255;
    int lane = w & 31, e2 = w >> 5;
    int e0 = e2 * 2;
    int ntiles = N >> 4;
    int kt = tile / ntiles, nt = tile - kt * ntiles;
    int k = kt * 32 + (e0 & 7) + ((lane >> 4) << 3) + ((e0 >> 3) << 4);
    int n = nt * 16 + (lane & 15);
    float a = W[(size_t)k * N + n];
    float b = W[(size_t)(k + 1) * N + n];
    ((unsigned int*)dst)[(size_t)tile * 256 + lane * 8 + e2] = pack2bf(a, b);
}

// ---------------------------------------------------------------------------
// Kernel 1: t = gelu(LN896(x @ W_in + b_in));  block = 16 rows x 896 cols
// ---------------------------------------------------------------------------
__global__ __launch_bounds__(256) void k_gemm1_ln_gelu(
    const float* __restrict__ x, const unsigned short* __restrict__ WinT,
    const float* __restrict__ bin, const float* __restrict__ g,
    const float* __restrict__ bb, float* __restrict__ out)
{
    __shared__ __align__(32) unsigned char smem[1024 + 57344 + 2048 + 128];
    unsigned short* sA  = (unsigned short*)smem;             // 1 A tile, frag order
    unsigned short* sB  = (unsigned short*)(smem + 1024);    // 56 B tiles, frag order
    float* rowbuf       = (float*)(smem + 1024);             // [16][896] f32 (reuse)
    float* pr           = (float*)(smem + 1024 + 57344);
    float* prs          = pr + 256;
    float* stat         = prs + 256;

    const int tid  = threadIdx.x;
    const int wave = tid >> 5;
    const int row0 = blockIdx.x * 16;

    v8f acc[7] = {};

    for (int k0 = 0; k0 < 2048; k0 += 32) {
        // B: 56 fragment tiles for this K-step = 3584 16B chunks (async copies)
        const unsigned short* src = WinT + (size_t)(k0 >> 5) * (56 * 512);
#pragma unroll
        for (int i = 0; i < 14; ++i) {
            int c = tid + i * 256;
            cp16(src + c * 8, sB + c * 8);
        }
        {                                                    // A: 16 rows x 32 k
            int r = tid >> 4, q = (tid & 15) * 2;
            const float2 v = *(const float2*)(x + (size_t)(row0 + r) * 2048 + k0 + q);
            stage_a1x2(sA, r, q, v.x, v.y);
        }
        cp_wait();
        __syncthreads();
        v16bf af = frag_ld(sA);
        v16bf bfr[7];
#pragma unroll
        for (int t = 0; t < 7; ++t) bfr[t] = frag_ld(sB + (wave * 7 + t) * 512);
#pragma unroll
        for (int t = 0; t < 7; ++t) acc[t] = wmma_bf16(af, bfr[t], acc[t]);
        __syncthreads();
    }

    // epilogue: +bias, stash f32 tile in LDS (aliases B stage)
    {
        int lane = tid & 31;
        int n16 = lane & 15, mh = (lane >> 4) * 8;
#pragma unroll
        for (int t = 0; t < 7; ++t) {
            int n = (wave * 7 + t) * 16 + n16;
            float bias = bin[n];
#pragma unroll
            for (int r = 0; r < 8; ++r)
                rowbuf[(mh + r) * 896 + n] = acc[t][r] + bias;
        }
    }
    __syncthreads();
    {
        int r = tid >> 4, j = tid & 15;
        float s = 0.f, ss = 0.f;
        const float* rp = rowbuf + r * 896 + j * 56;
#pragma unroll
        for (int i = 0; i < 56; ++i) { float v = rp[i]; s += v; ss += v * v; }
        pr[tid] = s; prs[tid] = ss;
    }
    __syncthreads();
    if (tid < 16) {
        float s = 0.f, ss = 0.f;
#pragma unroll
        for (int j = 0; j < 16; ++j) { s += pr[tid * 16 + j]; ss += prs[tid * 16 + j]; }
        float mean = s * (1.f / 896.f);
        float var  = ss * (1.f / 896.f) - mean * mean;
        stat[tid * 2]     = mean;
        stat[tid * 2 + 1] = rsqrtf(var + 1e-6f);
    }
    __syncthreads();
    for (int i = tid; i < 16 * 896; i += 256) {
        int r = i / 896, cidx = i - r * 896;
        float v = (rowbuf[i] - stat[r * 2]) * stat[r * 2 + 1] * g[cidx] + bb[cidx];
        float u = 0.7978845608028654f * (v + 0.044715f * v * v * v);
        out[(size_t)(row0 + r) * 896 + cidx] = 0.5f * v * (1.f + tanhf(u));
    }
}

// ---------------------------------------------------------------------------
// Kernel 2: per-line Fano GEMM: lineout[b,l,:] = w[l]*(triple_l @ W_l + b_l)
// grid = (B/32, 7); WT = pre-tiled [7*12 kt][8 nt] fragment tiles.
// ---------------------------------------------------------------------------
__global__ __launch_bounds__(256) void k_fano_line(
    const float* __restrict__ act,            // [B][896]
    const unsigned short* __restrict__ WT,    // tiled [84][8]*512
    const float* __restrict__ bias,           // [7][128]
    const float* __restrict__ wraw,           // [7]
    float* __restrict__ lineout)              // [B][7][128]
{
    __shared__ __align__(32) unsigned short sA[2 * 512];   // 2 A tiles
    __shared__ __align__(32) unsigned short sB[8 * 512];   // 8 B tiles
    const int tid  = threadIdx.x;
    const int wave = tid >> 5;
    const int l    = blockIdx.y;
    const int row0 = blockIdx.x * 32;

    float mx = wraw[0];
#pragma unroll
    for (int i = 1; i < 7; ++i) mx = fmaxf(mx, wraw[i]);
    float den = 0.f;
#pragma unroll
    for (int i = 0; i < 7; ++i) den += __expf(wraw[i] - mx);
    const float wl = __expf(wraw[l] - mx) / den;

    v8f acc[2] = {};
    const int ti0 = wave * 2;   // tiles: m-tile = ti>>3, n-tile = ti&7

    for (int ks = 0; ks < 12; ++ks) {
        int k0 = ks * 32;
        int p = c_fano[l][k0 >> 7];
        int colbase = k0 & 127;
        // B: 8 tiles = 512 chunks (async copies)
        const unsigned short* src = WT + ((size_t)(l * 12 + ks) * 8) * 512;
#pragma unroll
        for (int i = 0; i < 2; ++i) {
            int c = tid + i * 256;
            cp16(src + c * 8, sB + c * 8);
        }
        {   // A: 32 rows x 32 k, 1 float4/thread
            int r = tid >> 3, q = (tid & 7) * 4;
            const float4 v = *(const float4*)(act + (size_t)(row0 + r) * 896 + p * 128 + colbase + q);
            stage_a1x4(sA + (r >> 4) * 512, r & 15, q, v);
        }
        cp_wait();
        __syncthreads();
        v16bf af0 = frag_ld(sA + ((ti0 >> 3) & 1) * 512);
        v16bf b0  = frag_ld(sB + (ti0 & 7) * 512);
        v16bf b1  = frag_ld(sB + ((ti0 + 1) & 7) * 512);
        acc[0] = wmma_bf16(af0, b0, acc[0]);
        acc[1] = wmma_bf16(af0, b1, acc[1]);   // same m-tile (ti0 even)
        __syncthreads();
    }
    {
        int lane = tid & 31;
        int n16 = lane & 15, mh = (lane >> 4) * 8;
#pragma unroll
        for (int i = 0; i < 2; ++i) {
            int ti = ti0 + i;
            int n = (ti & 7) * 16 + n16;
            float bv = bias[l * 128 + n];
#pragma unroll
            for (int r = 0; r < 8; ++r) {
                int m = (ti >> 3) * 16 + mh + r;
                lineout[((size_t)(row0 + m) * 7 + l) * 128 + n] = (acc[i][r] + bv) * wl;
            }
        }
    }
}

// ---------------------------------------------------------------------------
// Kernel 3: scatter/3 + residual (each point p lies on lines c_fano[p]).
// ---------------------------------------------------------------------------
__global__ __launch_bounds__(256) void k_fano_combine(
    const float* __restrict__ tin, const float* __restrict__ lineout,
    float* __restrict__ cout)
{
    size_t idx4 = (size_t)blockIdx.x * 256 + threadIdx.x;     // over B*224 float4s
    if (idx4 >= (size_t)16384 * 224) return;
    size_t b = idx4 / 224;
    int rem = (int)(idx4 - b * 224);
    int p = rem >> 5;
    int j = (rem & 31) * 4;
    int l0 = c_fano[p][0], l1 = c_fano[p][1], l2 = c_fano[p][2];
    const float4 a = *(const float4*)(lineout + (b * 7 + l0) * 128 + j);
    const float4 c = *(const float4*)(lineout + (b * 7 + l1) * 128 + j);
    const float4 d = *(const float4*)(lineout + (b * 7 + l2) * 128 + j);
    const float4 t = *(const float4*)(tin + b * 896 + p * 128 + j);
    float4 o;
    o.x = t.x + (a.x + c.x + d.x) * (1.f / 3.f);
    o.y = t.y + (a.y + c.y + d.y) * (1.f / 3.f);
    o.z = t.z + (a.z + c.z + d.z) * (1.f / 3.f);
    o.w = t.w + (a.w + c.w + d.w) * (1.f / 3.f);
    *(float4*)(cout + b * 896 + p * 128 + j) = o;
}

// ---------------------------------------------------------------------------
// Kernel 4: z = c @ W_t2e + b;  8-level residual VQ over 240-word E8 codebook.
// ---------------------------------------------------------------------------
__global__ __launch_bounds__(256) void k_t2e_vq(
    const float* __restrict__ cin, const float* __restrict__ Wt2e,
    const float* __restrict__ bt2e, const float* __restrict__ codebook,
    float* __restrict__ zq)
{
    __shared__ float sW[896 * 8];
    __shared__ float sCB[240 * 8];
    __shared__ float sSQ[240];
    const int tid = threadIdx.x;
    for (int i = tid; i < 896 * 8; i += 256) sW[i] = Wt2e[i];
    for (int i = tid; i < 240 * 8; i += 256) sCB[i] = codebook[i];
    __syncthreads();
    for (int i = tid; i < 240; i += 256) {
        float s = 0.f;
#pragma unroll
        for (int o = 0; o < 8; ++o) { float v = sCB[i * 8 + o]; s += v * v; }
        sSQ[i] = s;
    }
    __syncthreads();

    const int lane = tid & 31;
    const size_t b = (size_t)blockIdx.x * 8 + (tid >> 5);
    const float* crow = cin + b * 896;

    float z[8] = {};
    for (int j = lane; j < 896; j += 32) {
        float cv = crow[j];
#pragma unroll
        for (int o = 0; o < 8; ++o) z[o] += cv * sW[j * 8 + o];
    }
#pragma unroll
    for (int m = 16; m >= 1; m >>= 1)
#pragma unroll
        for (int o = 0; o < 8; ++o) z[o] += __shfl_xor(z[o], m, 32);
#pragma unroll
    for (int o = 0; o < 8; ++o) z[o] += bt2e[o];

    float r[8];
#pragma unroll
    for (int o = 0; o < 8; ++o) r[o] = z[o];

    for (int lev = 0; lev < 8; ++lev) {
        float bd = 3.4e38f; int bi = 0x7fffffff;
        for (int i = lane; i < 240; i += 32) {
            float dp = 0.f;
#pragma unroll
            for (int o = 0; o < 8; ++o) dp += r[o] * sCB[i * 8 + o];
            float d = sSQ[i] - 2.f * dp;
            if (d < bd) { bd = d; bi = i; }
        }
#pragma unroll
        for (int m = 16; m >= 1; m >>= 1) {
            float od = __shfl_xor(bd, m, 32);
            int   oi = __shfl_xor(bi, m, 32);
            if (od < bd || (od == bd && oi < bi)) { bd = od; bi = oi; }
        }
#pragma unroll
        for (int o = 0; o < 8; ++o) r[o] -= sCB[bi * 8 + o];
    }
    if (lane < 8) {
        float v = 0.f;
#pragma unroll
        for (int o = 0; o < 8; ++o) if (o == lane) v = z[o] - r[o];
        zq[b * 8 + lane] = v;   // straight-through forward value == quantized
    }
}

// ---------------------------------------------------------------------------
// Kernel 5: t2 = zq @ W_e2t + b_e2t   (K=8, float4 over columns)
// ---------------------------------------------------------------------------
__global__ __launch_bounds__(256) void k_e2t(
    const float* __restrict__ zq, const float* __restrict__ We2t,
    const float* __restrict__ be2t, float* __restrict__ t2)
{
    size_t idx4 = (size_t)blockIdx.x * 256 + threadIdx.x;
    if (idx4 >= (size_t)16384 * 224) return;
    size_t b = idx4 / 224;
    int j = (int)(idx4 - b * 224) * 4;
    float zr[8];
#pragma unroll
    for (int k = 0; k < 8; ++k) zr[k] = zq[b * 8 + k];
    float4 acc = *(const float4*)(be2t + j);
#pragma unroll
    for (int k = 0; k < 8; ++k) {
        float4 w = *(const float4*)(We2t + k * 896 + j);
        acc.x += zr[k] * w.x; acc.y += zr[k] * w.y;
        acc.z += zr[k] * w.z; acc.w += zr[k] * w.w;
    }
    *(float4*)(t2 + b * 896 + j) = acc;
}

// ---------------------------------------------------------------------------
// Kernel 6: pre-LN output GEMM: out = c2 @ W_out + b_out   (grid.y = N/512)
// ---------------------------------------------------------------------------
__global__ __launch_bounds__(256) void k_gemm_out(
    const float* __restrict__ c2, const unsigned short* __restrict__ WoutT,
    const float* __restrict__ bout, float* __restrict__ out)
{
    __shared__ __align__(32) unsigned short sA[512];        // 1 A tile
    __shared__ __align__(32) unsigned short sB[32 * 512];   // 32 B tiles
    const int tid   = threadIdx.x;
    const int wave  = tid >> 5;
    const int row0  = blockIdx.x * 16;
    const int ncol0 = blockIdx.y * 512;

    v8f acc[4] = {};

    for (int k0 = 0; k0 < 896; k0 += 32) {
        // B: 32 fragment tiles = 2048 chunks (async copies)
        const unsigned short* src =
            WoutT + ((size_t)(k0 >> 5) * 128 + (ncol0 >> 4)) * 512;
#pragma unroll
        for (int i = 0; i < 8; ++i) {
            int c = tid + i * 256;
            cp16(src + c * 8, sB + c * 8);
        }
        {
            int r = tid >> 4, q = (tid & 15) * 2;
            const float2 v = *(const float2*)(c2 + (size_t)(row0 + r) * 896 + k0 + q);
            stage_a1x2(sA, r, q, v.x, v.y);
        }
        cp_wait();
        __syncthreads();
        v16bf af = frag_ld(sA);
        v16bf bfr[4];
#pragma unroll
        for (int t = 0; t < 4; ++t) bfr[t] = frag_ld(sB + (wave * 4 + t) * 512);
#pragma unroll
        for (int t = 0; t < 4; ++t) acc[t] = wmma_bf16(af, bfr[t], acc[t]);
        __syncthreads();
    }
    {
        int lane = tid & 31;
        int n16 = lane & 15, mh = (lane >> 4) * 8;
#pragma unroll
        for (int t = 0; t < 4; ++t) {
            int n = ncol0 + (wave * 4 + t) * 16 + n16;
            float bv = bout[n];
#pragma unroll
            for (int r = 0; r < 8; ++r)
                out[(size_t)(row0 + mh + r) * 2048 + n] = acc[t][r] + bv;
        }
    }
}

// ---------------------------------------------------------------------------
// Kernel 7: in-place LayerNorm over 2048 per row of d_out
// ---------------------------------------------------------------------------
__global__ __launch_bounds__(256) void k_ln2048(
    float* __restrict__ y, const float* __restrict__ g, const float* __restrict__ b)
{
    __shared__ float s1[256], s2[256];
    const int tid = threadIdx.x;
    float* row = y + (size_t)blockIdx.x * 2048;
    float s = 0.f, ss = 0.f;
#pragma unroll
    for (int i = 0; i < 8; ++i) {
        float v = row[tid + i * 256];
        s += v; ss += v * v;
    }
    s1[tid] = s; s2[tid] = ss;
    __syncthreads();
    for (int off = 128; off > 0; off >>= 1) {
        if (tid < off) { s1[tid] += s1[tid + off]; s2[tid] += s2[tid + off]; }
        __syncthreads();
    }
    float mean = s1[0] * (1.f / 2048.f);
    float var  = s2[0] * (1.f / 2048.f) - mean * mean;
    float rstd = rsqrtf(var + 1e-6f);
#pragma unroll
    for (int i = 0; i < 8; ++i) {
        int j = tid + i * 256;
        row[j] = (row[j] - mean) * rstd * g[j] + b[j];
    }
}

// ---------------------------------------------------------------------------
extern "C" void kernel_launch(void* const* d_in, const int* in_sizes, int n_in,
                              void* d_out, int out_size, void* d_ws, size_t ws_size,
                              hipStream_t stream)
{
    (void)in_sizes; (void)n_in; (void)out_size; (void)ws_size;
    const float* x     = (const float*)d_in[0];
    const float* W_in  = (const float*)d_in[1];
    const float* b_in  = (const float*)d_in[2];
    const float* ln1_g = (const float*)d_in[3];
    const float* ln1_b = (const float*)d_in[4];
    const float* lW1   = (const float*)d_in[5];
    const float* lb1   = (const float*)d_in[6];
    const float* lw1   = (const float*)d_in[7];
    const float* Wt2e  = (const float*)d_in[8];
    const float* bt2e  = (const float*)d_in[9];
    const float* cb    = (const float*)d_in[10];
    const float* We2t  = (const float*)d_in[11];
    const float* be2t  = (const float*)d_in[12];
    const float* lW2   = (const float*)d_in[13];
    const float* lb2   = (const float*)d_in[14];
    const float* lw2   = (const float*)d_in[15];
    const float* W_out = (const float*)d_in[16];
    const float* b_out = (const float*)d_in[17];
    const float* ln2_g = (const float*)d_in[18];
    const float* ln2_b = (const float*)d_in[19];
    float* out = (float*)d_out;

    float* buf0 = (float*)d_ws;                       // t / t2        [B][896]
    float* buf1 = buf0 + (size_t)16384 * 896;         // lineout       [B][7][128]
    float* buf2 = buf1 + (size_t)16384 * 896;         // c / c2        [B][896]
    float* zqb  = buf2 + (size_t)16384 * 896;         // zq            [B][8]
    unsigned short* WinT  = (unsigned short*)(zqb + (size_t)16384 * 8);
    unsigned short* WoutT = WinT  + (size_t)2048 * 896;   // bf16 frag tiles
    unsigned short* lWT1  = WoutT + (size_t)896 * 2048;
    unsigned short* lWT2  = lWT1  + (size_t)2688 * 128;

    const int nelem4 = (16384 * 224 + 255) / 256;

    // Pre-tile weights (bf16, fragment order)
    k_pretile<<<(2048 * 896 / 2 + 255) / 256, 256, 0, stream>>>(W_in,  WinT,  896,  2048 * 896 / 2);
    k_pretile<<<(896 * 2048 / 2 + 255) / 256, 256, 0, stream>>>(W_out, WoutT, 2048, 896 * 2048 / 2);
    k_pretile<<<(2688 * 128 / 2 + 255) / 256, 256, 0, stream>>>(lW1,   lWT1,  128,  2688 * 128 / 2);
    k_pretile<<<(2688 * 128 / 2 + 255) / 256, 256, 0, stream>>>(lW2,   lWT2,  128,  2688 * 128 / 2);

    k_gemm1_ln_gelu<<<1024, 256, 0, stream>>>(x, WinT, b_in, ln1_g, ln1_b, buf0);
    k_fano_line<<<dim3(512, 7), 256, 0, stream>>>(buf0, lWT1, lb1, lw1, buf1);
    k_fano_combine<<<nelem4, 256, 0, stream>>>(buf0, buf1, buf2);
    k_t2e_vq<<<2048, 256, 0, stream>>>(buf2, Wt2e, bt2e, cb, zqb);
    k_e2t<<<nelem4, 256, 0, stream>>>(zqb, We2t, be2t, buf0);
    k_fano_line<<<dim3(512, 7), 256, 0, stream>>>(buf0, lWT2, lb2, lw2, buf1);
    k_fano_combine<<<nelem4, 256, 0, stream>>>(buf0, buf1, buf2);
    k_gemm_out<<<dim3(1024, 4), 256, 0, stream>>>(buf2, WoutT, b_out, out);
    k_ln2048<<<16384, 256, 0, stream>>>(out, ln2_g, ln2_b);
}
